// myGCN_16690242912992
// MI455X (gfx1250) — compile-verified, matrix operand order checked by reading, not run on hardware
//
#include <hip/hip_runtime.h>

#define NNODES 50000
#define NEDGES 800000
#define FEAT   128
#define NTILES (NNODES / 16)   // 3125 exactly

typedef __attribute__((ext_vector_type(16))) __bf16    v16bf;
typedef __attribute__((ext_vector_type(8)))  float     v8f;
typedef __attribute__((ext_vector_type(4)))  unsigned  v4u;

// Fragment viewed either as the WMMA operand or as two 16-byte load payloads.
union Frag {
    v16bf v;
    v4u   q[2];
};

// ---------------------------------------------------------------- degree prep
__global__ void init_deg_kernel(float* __restrict__ deg) {
    int i = blockIdx.x * blockDim.x + threadIdx.x;
    if (i < NNODES) deg[i] = 1.0f;          // self-loop weight folded in
}

__global__ void accum_deg_kernel(const long long* __restrict__ col,
                                 const float* __restrict__ w,
                                 float* __restrict__ deg) {
    int e = blockIdx.x * blockDim.x + threadIdx.x;
    if (e < NEDGES) atomicAdd(&deg[(int)col[e]], w[e]);
}

__global__ void rsqrt_kernel(float* __restrict__ deg) {
    int i = blockIdx.x * blockDim.x + threadIdx.x;
    if (i < NNODES) {
        float d = deg[i];
        deg[i] = (d > 0.0f) ? rsqrtf(d) : 0.0f;   // deg >= 1 always, but match ref
    }
}

// ------------------------------------------------- feature / weight conversion
__global__ void cvt_x_kernel(const float* __restrict__ x,
                             float* __restrict__ out0,
                             __bf16* __restrict__ xb) {
    int i = blockIdx.x * blockDim.x + threadIdx.x;
    if (i < NNODES * FEAT) {
        float v = x[i];
        out0[i] = v;                 // embs[0] = input features (fp32)
        xb[i]   = (__bf16)v;         // bf16 working copy for WMMA
    }
}

// Store W transposed so B-fragment columns are contiguous in memory.
__global__ void cvt_wT_kernel(const float* __restrict__ W,
                              __bf16* __restrict__ WbT) {
    int i = blockIdx.x * blockDim.x + threadIdx.x;
    if (i < FEAT * FEAT) {
        int k = i >> 7, n = i & 127;
        WbT[n * FEAT + k] = (__bf16)W[i];
    }
}

// ------------------------------------------------------------------ WMMA GEMM
// h[N,128] = xb[N,128] @ W[128,128]  (W given transposed, bf16 in / f32 out)
// One wave per 16-row tile; 8 column tiles x 4 K-chunks = 32 wmma per wave.
__global__ void __launch_bounds__(128)
gemm_wmma_kernel(const __bf16* __restrict__ xb,
                 const __bf16* __restrict__ WbT,
                 float* __restrict__ h) {
    const int wave = threadIdx.x >> 5;
    const int lane = threadIdx.x & 31;
    const int tile = blockIdx.x * 4 + wave;
    if (tile >= NTILES) return;                 // uniform per wave: EXEC stays all-1s inside
    const int m0 = tile * 16;
    const int g  = lane >> 4;                   // half-wave group
    const int mr = lane & 15;                   // A row / B-C column within tile

    // A fragments for all 4 K-chunks (halves 0..7 <- K=8g+0..7, 8..15 <- K=16+8g+0..7)
    Frag a[4];
#pragma unroll
    for (int kc = 0; kc < 4; ++kc) {
        const __bf16* arow = xb + (size_t)(m0 + mr) * FEAT + kc * 32;
        a[kc].q[0] = *(const v4u*)(arow + 8 * g);
        a[kc].q[1] = *(const v4u*)(arow + 16 + 8 * g);
    }

#pragma unroll
    for (int nt = 0; nt < 8; ++nt) {
        v8f acc = {};
        const __bf16* bcol = WbT + (size_t)(nt * 16 + mr) * FEAT;
#pragma unroll
        for (int kc = 0; kc < 4; ++kc) {
            Frag b;
            b.q[0] = *(const v4u*)(bcol + kc * 32 + 8 * g);
            b.q[1] = *(const v4u*)(bcol + kc * 32 + 16 + 8 * g);
            acc = __builtin_amdgcn_wmma_f32_16x16x32_bf16(
                false, a[kc].v, false, b.v, (short)0, acc, false, false);
        }
        // C/D layout: VGPR r holds row (r + 8g), lane mr holds column nt*16+mr
        float* hrow = h + (size_t)(m0 + 8 * g) * FEAT + nt * 16 + mr;
#pragma unroll
        for (int r = 0; r < 8; ++r) hrow[(size_t)r * FEAT] = acc[r];
    }
}

// ---------------------------------------------------------------- aggregation
__global__ void zero_kernel(float* __restrict__ p, int n) {
    int i = blockIdx.x * blockDim.x + threadIdx.x;
    if (i < n) p[i] = 0.0f;
}

// One block per edge, one thread per feature; norm computed on the fly.
__global__ void scatter_kernel(const long long* __restrict__ row,
                               const long long* __restrict__ col,
                               const float* __restrict__ w,
                               const float* __restrict__ dis,
                               const float* __restrict__ h,
                               float* __restrict__ agg) {
    int e = blockIdx.x;
    int f = threadIdx.x;
    int r = (int)row[e];
    int c = (int)col[e];
    float nrm = dis[r] * w[e] * dis[c];
    atomicAdd(&agg[(size_t)c * FEAT + f], h[(size_t)r * FEAT + f] * nrm);
}

// Fused: self-loop message (h[i]*dis[i]^2) + bias + tanh.
// Writes fp32 output slab and bf16 features for the next layer.
__global__ void act_kernel(const float* __restrict__ agg,
                           const float* __restrict__ h,
                           const float* __restrict__ dis,
                           const float* __restrict__ bias,
                           float* __restrict__ out_slab,
                           __bf16* __restrict__ xb) {
    int i = blockIdx.x * blockDim.x + threadIdx.x;
    if (i >= NNODES * FEAT) return;
    int node = i >> 7;
    int f    = i & 127;
    float ds = dis[node];
    float v  = tanhf(agg[i] + h[i] * ds * ds + bias[f]);
    out_slab[i] = v;
    xb[i]       = (__bf16)v;
}

// -------------------------------------------------------------------- launch
extern "C" void kernel_launch(void* const* d_in, const int* in_sizes, int n_in,
                              void* d_out, int out_size, void* d_ws, size_t ws_size,
                              hipStream_t stream) {
    const float*     x   = (const float*)d_in[0];
    const long long* ei  = (const long long*)d_in[1];   // int64 [2, E]
    const float*     w   = (const float*)d_in[2];
    const float*     Wm[3]  = { (const float*)d_in[3], (const float*)d_in[5], (const float*)d_in[7] };
    const float*     bv[3]  = { (const float*)d_in[4], (const float*)d_in[6], (const float*)d_in[8] };
    const long long* row = ei;
    const long long* col = ei + NEDGES;
    float*           out = (float*)d_out;               // [4, N, 128] fp32

    // Workspace carve-out (256B aligned slots), ~64.3 MB total.
    char* p = (char*)d_ws;
    auto up = [](size_t v) { return (v + 255) & ~(size_t)255; };
    float*  dis = (float*)p;   p += up(sizeof(float)  * NNODES);
    float*  h   = (float*)p;   p += up(sizeof(float)  * NNODES * FEAT);
    float*  agg = (float*)p;   p += up(sizeof(float)  * NNODES * FEAT);
    __bf16* xb  = (__bf16*)p;  p += up(sizeof(__bf16) * NNODES * FEAT);
    __bf16* WbT = (__bf16*)p;  p += up(sizeof(__bf16) * FEAT * FEAT);

    const int T = 256;
    const int nf = NNODES * FEAT;

    // Normalization: deg (self-loops folded) -> dis = rsqrt(deg)
    init_deg_kernel <<<(NNODES + T - 1) / T, T, 0, stream>>>(dis);
    accum_deg_kernel<<<(NEDGES + T - 1) / T, T, 0, stream>>>(col, w, dis);
    rsqrt_kernel    <<<(NNODES + T - 1) / T, T, 0, stream>>>(dis);

    // embs[0] = x; bf16 working features
    cvt_x_kernel<<<(nf + T - 1) / T, T, 0, stream>>>(x, out, xb);

    for (int L = 0; L < 3; ++L) {
        cvt_wT_kernel<<<(FEAT * FEAT + T - 1) / T, T, 0, stream>>>(Wm[L], WbT);
        gemm_wmma_kernel<<<(NTILES + 3) / 4, 128, 0, stream>>>(xb, WbT, h);
        zero_kernel<<<(nf + T - 1) / T, T, 0, stream>>>(agg, nf);
        scatter_kernel<<<NEDGES, FEAT, 0, stream>>>(row, col, w, dis, h, agg);
        act_kernel<<<(nf + T - 1) / T, T, 0, stream>>>(
            agg, h, dis, bv[L], out + (size_t)(L + 1) * nf, xb);
    }
}
